// GAT_jmlr_64372969832706
// MI455X (gfx1250) — matrix operand order repaired and checked by reading, not compiled, not run on hardware
//
#include <hip/hip_runtime.h>
#include <hip/hip_bf16.h>
#include <math.h>

typedef __attribute__((ext_vector_type(16))) _Float16 v16h;
typedef __attribute__((ext_vector_type(8)))  float    v8f;

// ---------- float <-> monotone unsigned encoding for atomicMax on floats ----
__device__ __forceinline__ unsigned f2ord(float f) {
    unsigned u = __float_as_uint(f);
    return (u & 0x80000000u) ? ~u : (u | 0x80000000u);
}
__device__ __forceinline__ float ord2f(unsigned u) {
    return (u & 0x80000000u) ? __uint_as_float(u & 0x7FFFFFFFu)
                             : __uint_as_float(~u);
}

// ---------- zero mkey/denom [N] and agg [N*64] -------------------------------
__global__ void gat_zero_kernel(unsigned* __restrict__ mkey,
                                float* __restrict__ denom,
                                float* __restrict__ agg, int N) {
    size_t i = (size_t)blockIdx.x * blockDim.x + threadIdx.x;
    if (i < (size_t)N) { mkey[i] = 0u; denom[i] = 0.0f; }  // ord 0 == -max
    if (i < (size_t)N * 64) agg[i] = 0.0f;
}

// ---------- s[n] = rowsum(x[n,:]) / 8 ; one wave32 per node ------------------
__global__ void gat_rowsum_kernel(const float* __restrict__ x,
                                  float* __restrict__ s, int N) {
    int wid  = (int)(((size_t)blockIdx.x * blockDim.x + threadIdx.x) >> 5);
    int lane = threadIdx.x & 31;
    if (wid >= N) return;
    const float2* row = (const float2*)(x + (size_t)wid * 64);
    float2 v = row[lane];
    float t = v.x + v.y;
#pragma unroll
    for (int m = 16; m > 0; m >>= 1) t += __shfl_xor(t, m, 32);
    if (lane == 0) s[wid] = t * 0.125f;
}

// ---------- alpha_e = 4.5*(|sd+ss| - |sd-ss|); segment-max on src ------------
__global__ void gat_alpha_kernel(const long long* __restrict__ ei,
                                 const float* __restrict__ s,
                                 float* __restrict__ alpha,
                                 unsigned* __restrict__ mkey, int E, int N) {
    int e = blockIdx.x * blockDim.x + threadIdx.x;
    int ET = E + N;
    if (e >= ET) return;
    int src, dst;
    if (e < E) { src = (int)ei[e]; dst = (int)ei[(size_t)E + e]; }
    else       { src = dst = e - E; }
    float sd = s[dst], ss = s[src];
    float a = 4.5f * (fabsf(sd + ss) - fabsf(sd - ss));
    alpha[e] = a;
    atomicMax(mkey + src, f2ord(a));
}

// ---------- e = exp(alpha - m[src]); segment-sum on src ----------------------
__global__ void gat_expsum_kernel(const long long* __restrict__ ei,
                                  float* __restrict__ alpha,
                                  const unsigned* __restrict__ mkey,
                                  float* __restrict__ denom, int E, int N) {
    int e = blockIdx.x * blockDim.x + threadIdx.x;
    int ET = E + N;
    if (e >= ET) return;
    int src = (e < E) ? (int)ei[e] : (e - E);
    float ee = expf(alpha[e] - ord2f(mkey[src]));
    alpha[e] = ee;
    atomicAdd(denom + src, ee);
}

// ---------- agg[dst,:] += x[src,:] * (e/denom[src]); one wave per edge -------
__global__ void gat_aggregate_kernel(const long long* __restrict__ ei,
                                     const float* __restrict__ x,
                                     const float* __restrict__ ebuf,
                                     const float* __restrict__ denom,
                                     float* __restrict__ agg, int E, int N) {
    int wid  = (int)(((size_t)blockIdx.x * blockDim.x + threadIdx.x) >> 5);
    int lane = threadIdx.x & 31;
    int ET = E + N;
    if (wid >= ET) return;
    int src, dst;
    if (wid < E) { src = (int)ei[wid]; dst = (int)ei[(size_t)E + wid]; }
    else         { src = dst = wid - E; }
    const float2* xr = (const float2*)(x + (size_t)src * 64);
    __builtin_prefetch(xr + lane, 0, 3);                 // global_prefetch_b8
    float coeff = ebuf[wid] / (denom[src] + 1e-16f);
    float2 v = xr[lane];
    float* ag = agg + (size_t)dst * 64 + (size_t)lane * 2;
    atomicAdd(ag,     v.x * coeff);
    atomicAdd(ag + 1, v.y * coeff);
}

// ---------- WMMA fragment loaders (f32 memory -> f16 fragments) --------------
// A (16-bit 16x32): lane m<16 holds M=m, K=kc..kc+15;
// lanes 16..31 hold M=m, K=kc+16..kc+31.
__device__ __forceinline__ v16h load_a_frag(const float* __restrict__ A,
                                            int row, int kc, int hi) {
    const float* ar = A + (size_t)row * 64 + kc + (hi ? 16 : 0);
    v16h a;
#pragma unroll
    for (int j = 0; j < 8; ++j) a[j]     = (_Float16)ar[j];
#pragma unroll
    for (int j = 0; j < 8; ++j) a[8 + j] = (_Float16)ar[8 + j];
    return a;
}
// B (16-bit 32x16): lane n<16 holds N=n, K=kc..kc+15; upper half K=kc+16..kc+31.
// B[k][n] = W[n*64 + k]  (out = A @ W^T)
__device__ __forceinline__ v16h load_b_frag(const float* __restrict__ W,
                                            int col, int kc, int hi) {
    const float* wr = W + (size_t)col * 64 + kc + (hi ? 16 : 0);
    v16h b;
#pragma unroll
    for (int j = 0; j < 16; ++j) b[j] = (_Float16)wr[j];
    return b;
}

// ---------- H = relu(agg @ W1^T)  [Nx64]x[64x64] -----------------------------
// block = 128 threads = 4 waves; block tile = 16 rows x 64 cols (wave -> 16 cols)
__global__ void gat_gemm1_relu_kernel(const float* __restrict__ A,
                                      const float* __restrict__ W,
                                      float* __restrict__ H, int N) {
    int lane = threadIdx.x & 31;
    int wave = threadIdx.x >> 5;
    int r0 = blockIdx.x * 16;
    if (r0 >= N) return;                       // wave-uniform
    int colt = wave * 16;
    int m = lane & 15, hi = lane >> 4;
    int arow = r0 + m; if (arow >= N) arow = N - 1;   // clamp; rows >=N not stored
    v8f c = {};
#pragma unroll
    for (int kc = 0; kc < 64; kc += 32) {
        v16h a = load_a_frag(A, arow, kc, hi);
        v16h b = load_b_frag(W, colt + m, kc, hi);
        c = __builtin_amdgcn_wmma_f32_16x16x32_f16(false, a, false, b,
                                                   (short)0, c, false, false);
    }
    // D layout: VGPR j -> row r0 + j + 8*hi, col colt + (lane&15)
    float* Hbase = H + (size_t)(r0 + 8 * hi) * 64 + colt + m;
    if (r0 + 16 <= N) {                        // uniform full-tile fast path
#pragma unroll
        for (int j = 0; j < 8; ++j) Hbase[(size_t)j * 64] = fmaxf(c[j], 0.0f);
    } else {
#pragma unroll
        for (int j = 0; j < 8; ++j)
            if (r0 + j + 8 * hi < N) Hbase[(size_t)j * 64] = fmaxf(c[j], 0.0f);
    }
}

// ---------- out = log_softmax(agg @ W2^T)  [Nx64]x[64x16] --------------------
// block = 128 threads = 4 waves; each wave -> one 16x16 tile (16 rows)
__global__ void gat_gemm2_lsm_kernel(const float* __restrict__ A,
                                     const float* __restrict__ W,
                                     float* __restrict__ out, int N) {
    int lane = threadIdx.x & 31;
    int wave = threadIdx.x >> 5;
    int tile = blockIdx.x * 4 + wave;
    int r0 = tile * 16;
    if (r0 >= N) return;                       // wave-uniform
    int m = lane & 15, hi = lane >> 4;
    int arow = r0 + m; if (arow >= N) arow = N - 1;
    v8f c = {};
#pragma unroll
    for (int kc = 0; kc < 64; kc += 32) {
        v16h a = load_a_frag(A, arow, kc, hi);
        v16h b = load_b_frag(W, m, kc, hi);    // 16 output cols
        c = __builtin_amdgcn_wmma_f32_16x16x32_f16(false, a, false, b,
                                                   (short)0, c, false, false);
    }
    // log_softmax across the 16 columns of each row: row j(+8*hi) lives in
    // VGPR j across a 16-lane group -> reduce with shfl_xor width 16.
    v8f r;
#pragma unroll
    for (int j = 0; j < 8; ++j) {
        float v = c[j];
        float mx = v;
#pragma unroll
        for (int msk = 1; msk < 16; msk <<= 1) mx = fmaxf(mx, __shfl_xor(mx, msk, 16));
        float e = expf(v - mx);
        float sum = e;
#pragma unroll
        for (int msk = 1; msk < 16; msk <<= 1) sum += __shfl_xor(sum, msk, 16);
        r[j] = (v - mx) - logf(sum);
    }
    float* Obase = out + (size_t)(r0 + 8 * hi) * 16 + m;
    if (r0 + 16 <= N) {                        // uniform full-tile fast path
#pragma unroll
        for (int j = 0; j < 8; ++j) Obase[(size_t)j * 16] = r[j];
    } else {
#pragma unroll
        for (int j = 0; j < 8; ++j)
            if (r0 + j + 8 * hi < N) Obase[(size_t)j * 16] = r[j];
    }
}

// ---------------------------------------------------------------------------
extern "C" void kernel_launch(void* const* d_in, const int* in_sizes, int n_in,
                              void* d_out, int out_size, void* d_ws, size_t ws_size,
                              hipStream_t stream) {
    const float*     x  = (const float*)d_in[0];
    const long long* ei = (const long long*)d_in[1];   // int64 edge_index [2,E]
    const float*     W1 = (const float*)d_in[2];       // [64,64]
    const float*     W2 = (const float*)d_in[3];       // [16,64]
    float*           out = (float*)d_out;

    const int N  = in_sizes[0] / 64;
    const int E  = in_sizes[1] / 2;
    const int ET = E + N;

    auto au = [](size_t v) { return (v + 255) & ~(size_t)255; };
    char* ws = (char*)d_ws;
    size_t off = 0;
    float*    s     = (float*)(ws + off);    off += au((size_t)N * 4);
    unsigned* mkey  = (unsigned*)(ws + off); off += au((size_t)N * 4);
    float*    denom = (float*)(ws + off);    off += au((size_t)N * 4);
    float*    alpha = (float*)(ws + off);    off += au((size_t)ET * 4);
    float*    agg   = (float*)(ws + off);    off += au((size_t)N * 64 * 4);
    float*    h     = (float*)(ws + off);    off += au((size_t)N * 64 * 4);
    (void)ws_size; (void)n_in; (void)out_size;

    const int TB = 256;
    dim3 b256(TB);
    dim3 gZero((unsigned)(((size_t)N * 64 + TB - 1) / TB));
    dim3 gNodeW((N * 32 + TB - 1) / TB);          // wave per node
    dim3 gEdge((ET + TB - 1) / TB);               // thread per edge
    dim3 gEdgeW((unsigned)(((size_t)ET * 32 + TB - 1) / TB)); // wave per edge
    dim3 gGemm1((N + 15) / 16);
    dim3 gGemm2(((N + 15) / 16 + 3) / 4);

    // ----- layer 1 -----
    gat_zero_kernel<<<gZero, b256, 0, stream>>>(mkey, denom, agg, N);
    gat_rowsum_kernel<<<gNodeW, b256, 0, stream>>>(x, s, N);
    gat_alpha_kernel<<<gEdge, b256, 0, stream>>>(ei, s, alpha, mkey, E, N);
    gat_expsum_kernel<<<gEdge, b256, 0, stream>>>(ei, alpha, mkey, denom, E, N);
    gat_aggregate_kernel<<<gEdgeW, b256, 0, stream>>>(ei, x, alpha, denom, agg, E, N);
    gat_gemm1_relu_kernel<<<gGemm1, dim3(128), 0, stream>>>(agg, W1, h, N);

    // ----- layer 2 -----
    gat_zero_kernel<<<gZero, b256, 0, stream>>>(mkey, denom, agg, N);
    gat_rowsum_kernel<<<gNodeW, b256, 0, stream>>>(h, s, N);
    gat_alpha_kernel<<<gEdge, b256, 0, stream>>>(ei, s, alpha, mkey, E, N);
    gat_expsum_kernel<<<gEdge, b256, 0, stream>>>(ei, alpha, mkey, denom, E, N);
    gat_aggregate_kernel<<<gEdgeW, b256, 0, stream>>>(ei, h, alpha, denom, agg, E, N);
    gat_gemm2_lsm_kernel<<<gGemm2, dim3(128), 0, stream>>>(agg, W2, out, N);
}
